// TymysLLM_46840913330197
// MI455X (gfx1250) — compile-verified, matrix operand order checked by reading, not run on hardware
//
#include <hip/hip_runtime.h>
#include <hip/hip_bf16.h>
#include <math.h>

// ---------------------------------------------------------------------------
// TymysLLM block for MI455X (gfx1250, wave32, WMMA).
// All GEMMs run through v_wmma_f32_16x16x32_bf16 (f32 accumulate).
// ---------------------------------------------------------------------------

typedef __attribute__((ext_vector_type(16))) __bf16 v16bf;
typedef __attribute__((ext_vector_type(8)))  float  v8f;

#define TILE_M 128
#define TILE_N 128
#define TILE_K 32

#define LN_EPS 1e-5f
#define BN_EPS 1e-5f

// Optional CDNA5 async global->LDS copy path (ASYNCcnt-tracked).
#if defined(__has_builtin)
#  if __has_builtin(__builtin_amdgcn_global_load_async_to_lds_b128)
#    define HAVE_ASYNC_LDS 1
#  endif
#  if __has_builtin(__builtin_amdgcn_s_wait_asynccnt)
#    define HAVE_WAIT_ASYNC_BUILTIN 1
#  endif
#endif
#ifndef HAVE_ASYNC_LDS
#  define HAVE_ASYNC_LDS 0
#endif
#ifndef HAVE_WAIT_ASYNC_BUILTIN
#  define HAVE_WAIT_ASYNC_BUILTIN 0
#endif

#if HAVE_ASYNC_LDS
// The builtin takes (global int4*, lds int4*, imm offset, imm cpol); clang
// prints the pointee as vector_size(16) int with AS1 ("__device__").
typedef int v4i_ls __attribute__((vector_size(16)));
typedef __attribute__((address_space(1))) v4i_ls* as1_v4i;
typedef __attribute__((address_space(3))) v4i_ls* as3_v4i;

__device__ __forceinline__ void async_copy_b128(const void* g, void* l) {
    __builtin_amdgcn_global_load_async_to_lds_b128((as1_v4i)(g), (as3_v4i)(l), 0, 0);
}
__device__ __forceinline__ void wait_async0() {
#  if HAVE_WAIT_ASYNC_BUILTIN
    __builtin_amdgcn_s_wait_asynccnt(0);
#  else
    asm volatile("s_wait_asynccnt 0x0" ::: "memory");
#  endif
}
#endif

// round-to-nearest-even f32 -> bf16 bits
__device__ __forceinline__ unsigned short f2bf(float f) {
    unsigned int u = __float_as_uint(f);
    unsigned int r = u + 0x7FFFu + ((u >> 16) & 1u);
    return (unsigned short)(r >> 16);
}

__device__ __forceinline__ float softplusf(float x) {
    return fmaxf(x, 0.f) + log1pf(expf(-fabsf(x)));
}

// ---------------------------------------------------------------------------
// Generic bf16 GEMM: C[M,N] = A[M,K] * B[K,N] (+bias) (+relu)
// Block: 256 threads = 8 waves; block tile 128x128; BK=32 (one WMMA K step).
// Wave (wm in 0..1, wn in 0..3) owns a 64x32 tile = 4x2 fragments of 16x16.
// Double-buffered LDS; A staged via async-to-LDS (if available), B staged via
// coalesced uint4 loads + transposed [n][k] stores so fragment K-pairs are
// two contiguous ds_load_b128 per fragment, matching the 16-bit A/B layout:
//   lane 0-15 -> row M (or col N), lane 16-31 -> same row, K group +8
//   pair p<4: K=2p ; p>=4: K=2(p-4)+16 ; +8 if lane>=16
// ---------------------------------------------------------------------------
union FragAB { v16bf v; unsigned int u[8]; };
union Pack8  { uint4 q; unsigned short s[8]; };

__global__ __launch_bounds__(256) void gemm_bf16_kernel(
    const unsigned short* __restrict__ A, const unsigned short* __restrict__ B,
    const float* __restrict__ bias, float* __restrict__ Cf,
    unsigned short* __restrict__ Cb, int M, int N, int K, int relu)
{
    __shared__ unsigned short As[2][TILE_M][TILE_K];
    __shared__ unsigned short Bs[2][TILE_N][TILE_K];   // transposed: [n][k]

    const int tid  = threadIdx.x;
    const int lane = tid & 31;
    const int wave = tid >> 5;
    const int wm   = wave >> 2;     // 0..1 -> 64 rows
    const int wn   = wave & 3;      // 0..3 -> 32 cols
    const int r    = lane & 15;
    const int half = lane >> 4;

    const int blockM = blockIdx.y * TILE_M;
    const int blockN = blockIdx.x * TILE_N;

    v8f acc[4][2];
#pragma unroll
    for (int mt = 0; mt < 4; ++mt)
#pragma unroll
        for (int nt = 0; nt < 2; ++nt)
#pragma unroll
            for (int i = 0; i < 8; ++i) acc[mt][nt][i] = 0.f;

    // Per-thread staging assignments (hoisted; advance by constants in loop).
    const int arow  = tid >> 1;            // 2 threads per A row
    const int aseg  = (tid & 1) * 16;      // each covers 16 of 32 K elems
    const int brow  = tid >> 3;            // 0..31  (K within tile)
    const int bcol0 = (tid & 7) * 16;      // 0..112 (N within tile)

    const unsigned short* aptr = A + (size_t)(blockM + arow) * K + aseg;
    const unsigned short* bptr = B + (size_t)brow * N + blockN + bcol0;
    const size_t bstep = (size_t)TILE_K * N;

    auto stage = [&](int buf) {
        // ---- A tile: 2 x 16B per thread ----
#if HAVE_ASYNC_LDS
        async_copy_b128(aptr,     &As[buf][arow][aseg]);
        async_copy_b128(aptr + 8, &As[buf][arow][aseg + 8]);
#else
        const uint4* asrc = reinterpret_cast<const uint4*>(aptr);
        *reinterpret_cast<uint4*>(&As[buf][arow][aseg])     = asrc[0];
        *reinterpret_cast<uint4*>(&As[buf][arow][aseg + 8]) = asrc[1];
#endif
        // ---- B tile: coalesced 2 x 16B loads, transposed stores ----
        Pack8 b0, b1;
        b0.q = reinterpret_cast<const uint4*>(bptr)[0];
        b1.q = reinterpret_cast<const uint4*>(bptr)[1];
#pragma unroll
        for (int j = 0; j < 8; ++j) Bs[buf][bcol0 + j][brow]     = b0.s[j];
#pragma unroll
        for (int j = 0; j < 8; ++j) Bs[buf][bcol0 + 8 + j][brow] = b1.s[j];
        aptr += TILE_K;
        bptr += bstep;
    };

    const int nk = K / TILE_K;
    stage(0);
#if HAVE_ASYNC_LDS
    wait_async0();
#endif
    __syncthreads();

    for (int it = 0; it < nk; ++it) {
        const int cur = it & 1;
        if (it + 1 < nk) stage(cur ^ 1);

        // --- build fragments from LDS ---
        FragAB af[4], bfr[2];
#pragma unroll
        for (int mt = 0; mt < 4; ++mt) {
            const int row = wm * 64 + mt * 16 + r;
#pragma unroll
            for (int p = 0; p < 8; ++p) {
                const int kk = ((p < 4) ? 2 * p : 2 * (p - 4) + 16) + half * 8;
                af[mt].u[p] = *reinterpret_cast<const unsigned int*>(&As[cur][row][kk]);
            }
        }
#pragma unroll
        for (int nt = 0; nt < 2; ++nt) {
            const int col = wn * 32 + nt * 16 + r;
#pragma unroll
            for (int p = 0; p < 8; ++p) {
                const int kk = ((p < 4) ? 2 * p : 2 * (p - 4) + 16) + half * 8;
                bfr[nt].u[p] = *reinterpret_cast<const unsigned int*>(&Bs[cur][col][kk]);
            }
        }

        // --- 8 WMMAs per wave per K step ---
#pragma unroll
        for (int mt = 0; mt < 4; ++mt)
#pragma unroll
            for (int nt = 0; nt < 2; ++nt)
                acc[mt][nt] = __builtin_amdgcn_wmma_f32_16x16x32_bf16(
                    false, af[mt].v, false, bfr[nt].v,
                    (short)0, acc[mt][nt], false, false);

#if HAVE_ASYNC_LDS
        wait_async0();     // our async stores to the other buffer are done
#endif
        __syncthreads();   // everyone done reading cur + staging next
    }

    // --- epilogue: C layout VGPR i -> M = i + 8*half, N = lane&15 ---
#pragma unroll
    for (int mt = 0; mt < 4; ++mt)
#pragma unroll
        for (int nt = 0; nt < 2; ++nt) {
            const int col = blockN + wn * 32 + nt * 16 + r;
            const float bv = bias ? bias[col] : 0.f;
#pragma unroll
            for (int i = 0; i < 8; ++i) {
                const int row = blockM + wm * 64 + mt * 16 + half * 8 + i;
                float v = acc[mt][nt][i] + bv;
                if (relu) v = fmaxf(v, 0.f);
                const size_t idx = (size_t)row * N + col;
                if (Cf) Cf[idx] = v;
                if (Cb) Cb[idx] = f2bf(v);
            }
        }
}

// ---------------------------------------------------------------------------
// Elementwise / support kernels
// ---------------------------------------------------------------------------
__global__ void f32_to_bf16_kernel(const float* __restrict__ src,
                                   unsigned short* __restrict__ dst, int n) {
    for (int i = blockIdx.x * blockDim.x + threadIdx.x; i < n;
         i += gridDim.x * blockDim.x)
        dst[i] = f2bf(src[i]);
}

// conv_w [O=D, I=D, W=3] (OIW) -> B matrix [3*D (k-major, then i), D] bf16
__global__ void pack_conv_kernel(const float* __restrict__ w,
                                 unsigned short* __restrict__ out, int D) {
    const int total = D * D * 3;
    for (int idx = blockIdx.x * blockDim.x + threadIdx.x; idx < total;
         idx += gridDim.x * blockDim.x) {
        const int o = idx / (D * 3);
        const int rem = idx - o * (D * 3);
        const int i = rem / 3;
        const int k = rem - i * 3;
        out[(size_t)(k * D + i) * D + o] = f2bf(w[idx]);
    }
}

__global__ void embed_kernel(const int* __restrict__ X, const float* __restrict__ emb,
                             float* __restrict__ xf, unsigned short* __restrict__ xb,
                             int D) {
    const int token = blockIdx.x;
    const size_t rowoff = (size_t)X[token] * D;
    for (int j = threadIdx.x; j < D; j += blockDim.x) {
        const float v = emb[rowoff + j];
        xf[(size_t)token * D + j] = v;
        xb[(size_t)token * D + j] = f2bf(v);
    }
}

// minGRU Heinsen scan (streaming log-cumsum-exp), one thread per (b, d).
__global__ void mingru_scan_kernel(const float* __restrict__ hg,
                                   float* __restrict__ h, int Bn, int S, int D) {
    const int t = blockIdx.x * blockDim.x + threadIdx.x;
    if (t >= Bn * D) return;
    const int b = t / D;
    const int d = t - b * D;
    float a = 0.f;          // cumsum(log_coeffs)
    float lse = -INFINITY;  // running logcumsumexp(log_values - a_star)
    for (int s = 0; s < S; ++s) {
        const size_t base = ((size_t)(b * S + s)) * (2 * D);
        const float hid  = hg[base + d];
        const float gate = hg[base + D + d];
        const float log_coeff = -softplusf(gate);                  // log(1-z)
        const float lg = (hid >= 0.f) ? logf(hid + 0.5f) : -softplusf(-hid);
        const float log_val = -softplusf(-gate) + lg;              // log z + log h~
        a += log_coeff;
        const float v = log_val - a;
        if (v > lse) lse = v + log1pf(expf(lse - v));
        else         lse = lse + log1pf(expf(v - lse));
        h[((size_t)(b * S + s)) * D + d] = expf(a + lse);
    }
}

// LayerNorm over D + residual add; one block per token.
__global__ __launch_bounds__(256) void ln_residual_kernel(
    const float* __restrict__ h, const float* __restrict__ shortcut,
    const float* __restrict__ g, const float* __restrict__ b,
    float* __restrict__ out, int D)
{
    __shared__ float r1[256], r2[256];
    const int token = blockIdx.x;
    const size_t base = (size_t)token * D;
    float hv[4];
    float s1 = 0.f, s2 = 0.f;
    int c = 0;
    for (int j = threadIdx.x; j < D; j += 256, ++c) {
        const float v = h[base + j];
        hv[c] = v; s1 += v; s2 += v * v;
    }
    r1[threadIdx.x] = s1; r2[threadIdx.x] = s2;
    __syncthreads();
    for (int o = 128; o > 0; o >>= 1) {
        if (threadIdx.x < o) {
            r1[threadIdx.x] += r1[threadIdx.x + o];
            r2[threadIdx.x] += r2[threadIdx.x + o];
        }
        __syncthreads();
    }
    if (threadIdx.x == 0) {
        const float mu  = r1[0] / D;
        const float var = r2[0] / D - mu * mu;
        r1[0] = mu;
        r2[0] = rsqrtf(var + LN_EPS);
    }
    __syncthreads();
    const float mu = r1[0], rstd = r2[0];
    c = 0;
    for (int j = threadIdx.x; j < D; j += 256, ++c)
        out[base + j] = (hv[c] - mu) * rstd * g[j] + b[j] + shortcut[base + j];
}

// im2col for conv1d k=3 pad=1 along S: out[token, k*D+i] = x[b, s+k-1, i]
__global__ void im2col_kernel(const float* __restrict__ x,
                              unsigned short* __restrict__ out, int S, int D) {
    const int token = blockIdx.x;
    const int b = token / S;
    const int s = token - b * S;
    for (int j = threadIdx.x; j < 3 * D; j += blockDim.x) {
        const int k = j / D;
        const int i = j - k * D;
        const int sp = s + k - 1;
        float v = 0.f;
        if (sp >= 0 && sp < S) v = x[((size_t)(b * S + sp)) * D + i];
        out[(size_t)token * (3 * D) + j] = f2bf(v);
    }
}

// BatchNorm training-mode stats per channel over all tokens; one block / chan.
__global__ __launch_bounds__(256) void bn_stats_kernel(
    const float* __restrict__ xc, const float* __restrict__ g,
    const float* __restrict__ b, float* __restrict__ scale,
    float* __restrict__ shift, int NT, int D)
{
    __shared__ float r1[256], r2[256];
    const int c = blockIdx.x;
    float s1 = 0.f, s2 = 0.f;
    for (int t = threadIdx.x; t < NT; t += 256) {
        const float v = xc[(size_t)t * D + c];
        s1 += v; s2 += v * v;
    }
    r1[threadIdx.x] = s1; r2[threadIdx.x] = s2;
    __syncthreads();
    for (int o = 128; o > 0; o >>= 1) {
        if (threadIdx.x < o) {
            r1[threadIdx.x] += r1[threadIdx.x + o];
            r2[threadIdx.x] += r2[threadIdx.x + o];
        }
        __syncthreads();
    }
    if (threadIdx.x == 0) {
        const float mu  = r1[0] / NT;
        const float var = r2[0] / NT - mu * mu;
        const float sc  = g[c] * rsqrtf(var + BN_EPS);
        scale[c] = sc;
        shift[c] = b[c] - mu * sc;
    }
}

__global__ void bn_apply_kernel(const float* __restrict__ xc,
                                const float* __restrict__ scale,
                                const float* __restrict__ shift,
                                unsigned short* __restrict__ out, int total, int D) {
    for (int i = blockIdx.x * blockDim.x + threadIdx.x; i < total;
         i += gridDim.x * blockDim.x) {
        const int d = i % D;
        out[i] = f2bf(xc[i] * scale[d] + shift[d]);
    }
}

// ---------------------------------------------------------------------------
// Launch
// ---------------------------------------------------------------------------
extern "C" void kernel_launch(void* const* d_in, const int* in_sizes, int n_in,
                              void* d_out, int out_size, void* d_ws, size_t ws_size,
                              hipStream_t stream) {
    (void)in_sizes; (void)n_in; (void)out_size; (void)ws_size;
    const int Bn = 2, S = 2048, D = 1024, V = 32000;
    const int NT = Bn * S;       // 4096 tokens
    const int D2 = 2 * D;        // 2048
    const int K3 = 3 * D;        // 3072

    const int*   X      = (const int*)  d_in[0];
    const float* emb    = (const float*)d_in[1];
    const float* w_hg   = (const float*)d_in[2];
    const float* conv_w = (const float*)d_in[3];
    const float* conv_b = (const float*)d_in[4];
    const float* bn_g   = (const float*)d_in[5];
    const float* bn_b   = (const float*)d_in[6];
    const float* ln_g   = (const float*)d_in[7];
    const float* ln_b   = (const float*)d_in[8];
    const float* w1     = (const float*)d_in[9];
    const float* b1     = (const float*)d_in[10];
    const float* w2     = (const float*)d_in[11];
    const float* b2     = (const float*)d_in[12];
    const float* w3     = (const float*)d_in[13];
    const float* b3     = (const float*)d_in[14];
    float* out = (float*)d_out;

    // -------- workspace carve-up (buffers reused across phases) --------
    char* ws = (char*)d_ws;
    size_t off = 0;
    auto alloc = [&](size_t bytes) -> void* {
        void* p = ws + off;
        off = (off + bytes + 255) & ~(size_t)255;
        return p;
    };
    // bufA: x_f32 (shortcut) -> conv output f32
    float* bufA = (float*)alloc((size_t)NT * D * 4);
    // bufB: x bf16 (GEMM1 A) -> BN output bf16 (w1 A)
    unsigned short* bufB = (unsigned short*)alloc((size_t)NT * D * 2);
    // bufC: hg f32 [NT,2D] -> im2col bf16 [NT,3D]
    void* bufC = alloc((size_t)NT * D2 * 4);
    // bufD: h f32 [NT,D] -> act1 bf16 [NT,2D]
    void* bufD = alloc((size_t)NT * D * 4);
    // bufE: xln f32 [NT,D] -> act2 bf16 [NT,2D]
    void* bufE = alloc((size_t)NT * D * 4);
    float* bn_scale = (float*)alloc((size_t)D * 4);
    float* bn_shift = (float*)alloc((size_t)D * 4);
    unsigned short* wb_hg   = (unsigned short*)alloc((size_t)D * D2 * 2);
    unsigned short* wb_conv = (unsigned short*)alloc((size_t)K3 * D * 2);
    unsigned short* wb1     = (unsigned short*)alloc((size_t)D * D2 * 2);
    unsigned short* wb2     = (unsigned short*)alloc((size_t)D2 * D2 * 2);
    unsigned short* wb3     = (unsigned short*)alloc((size_t)D2 * V * 2);

    auto gemm = [&](const unsigned short* A, const unsigned short* Bm,
                    const float* bias, float* Cf, unsigned short* Cb,
                    int M, int N, int K, int relu) {
        dim3 grid(N / TILE_N, M / TILE_M);
        gemm_bf16_kernel<<<grid, 256, 0, stream>>>(A, Bm, bias, Cf, Cb, M, N, K, relu);
    };

    // -------- weight conversion (f32 -> bf16) --------
    f32_to_bf16_kernel<<<1024, 256, 0, stream>>>(w_hg, wb_hg, D * D2);
    pack_conv_kernel  <<<1024, 256, 0, stream>>>(conv_w, wb_conv, D);
    f32_to_bf16_kernel<<<1024, 256, 0, stream>>>(w1, wb1, D * D2);
    f32_to_bf16_kernel<<<1024, 256, 0, stream>>>(w2, wb2, D2 * D2);
    f32_to_bf16_kernel<<<2048, 256, 0, stream>>>(w3, wb3, D2 * V);

    // -------- forward pipeline --------
    // x = emb[X]
    embed_kernel<<<NT, 256, 0, stream>>>(X, emb, bufA, bufB, D);
    // hg = x @ w_hg   [4096,1024] x [1024,2048] -> f32
    gemm(bufB, wb_hg, nullptr, (float*)bufC, nullptr, NT, D2, D, 0);
    // h = minGRU scan
    mingru_scan_kernel<<<(Bn * D + 255) / 256, 256, 0, stream>>>(
        (const float*)bufC, (float*)bufD, Bn, S, D);
    // x = LN(h) + shortcut
    ln_residual_kernel<<<NT, 256, 0, stream>>>(
        (const float*)bufD, bufA, ln_g, ln_b, (float*)bufE, D);
    // im2col (k=3, pad=1) -> bf16 [4096, 3072]
    im2col_kernel<<<NT, 256, 0, stream>>>((const float*)bufE,
                                          (unsigned short*)bufC, S, D);
    // conv as GEMM: [4096,3072] x [3072,1024] + conv_b -> f32
    gemm((const unsigned short*)bufC, wb_conv, conv_b, bufA, nullptr, NT, D, K3, 0);
    // BN (training-mode stats over B,S)
    bn_stats_kernel<<<D, 256, 0, stream>>>(bufA, bn_g, bn_b, bn_scale, bn_shift, NT, D);
    bn_apply_kernel<<<2048, 256, 0, stream>>>(bufA, bn_scale, bn_shift, bufB, NT * D, D);
    // x = relu(x @ w1 + b1)   -> bf16 [4096,2048]
    gemm(bufB, wb1, b1, nullptr, (unsigned short*)bufD, NT, D2, D, 1);
    // x = relu(x @ w2 + b2)   -> bf16 [4096,2048]
    gemm((const unsigned short*)bufD, wb2, b2, nullptr, (unsigned short*)bufE, NT, D2, D2, 1);
    // logits = x @ w3 + b3    -> f32 [4096,32000] (covers all of d_out)
    gemm((const unsigned short*)bufE, wb3, b3, out, nullptr, NT, V, D2, 0);
}
